// ExpWindowAttention_13735305413419
// MI455X (gfx1250) — matrix operand order repaired and checked by reading, not compiled, not run on hardware
//
#include <hip/hip_runtime.h>
#include <hip/hip_bf16.h>
#include <math.h>

// ---------------------------------------------------------------------------
// ExpWindowAttention for MI455X (gfx1250, wave32, WMMA).
//
// Shapes: B=32, L=2048, D=1024, H=8, hd=128, W=5, T=11, Nw=409, nC=2.
// Strategy: per-POSITION K/V projection (2.2x fewer FLOPs than per-window),
// all D x D GEMMs via v_wmma_f32_16x16x32_bf16 (f32 accumulate). The K-loop
// is unrolled x2 with ping-pong fragment buffers: loads for step k+32/k+64
// overlap the 4 WMMAs of step k, and the two buffers alternate structurally
// so no register-copy rotation is needed. Tiny attention/softmax/classifier
// stages run as wave32 VALU kernels.
// ---------------------------------------------------------------------------

typedef __attribute__((ext_vector_type(16))) __bf16 v16bf;
typedef __attribute__((ext_vector_type(8)))  __bf16 v8bf;
typedef __attribute__((ext_vector_type(4)))  __bf16 v4bf;
typedef __attribute__((ext_vector_type(8)))  float  v8f;

#define B_   32
#define L_   2048
#define D_   1024
#define H_   8
#define HD_  128
#define W_   5
#define T_   11
#define NW_  409          // number of windows (starts 0,5,...,2040)
#define NWP_ 416          // NW padded to multiple of 16 (GEMM tiling)
#define LP_  2064         // L+pad rows, multiple of 16 (covers pos 0..2050)
#define NC_  2

// ---------------------------------------------------------------------------
// fp32 -> bf16 cast (weights)
// ---------------------------------------------------------------------------
__global__ void cvt_f32_bf16(const float* __restrict__ in,
                             __bf16* __restrict__ out, size_t n) {
  size_t i = (size_t)blockIdx.x * blockDim.x + threadIdx.x;
  if (i < n) out[i] = (__bf16)in[i];
}

// ---------------------------------------------------------------------------
// Build zero-padded bf16 copy of x: [B, LP_, D]  (rows l>=L are zero)
// ---------------------------------------------------------------------------
__global__ void build_xpad(const float* __restrict__ x, __bf16* __restrict__ xp) {
  size_t i = (size_t)blockIdx.x * blockDim.x + threadIdx.x;
  const size_t total = (size_t)B_ * LP_ * D_;
  if (i >= total) return;
  int d = (int)(i & (D_ - 1));
  size_t row = i >> 10;
  int l = (int)(row % LP_);
  int b = (int)(row / LP_);
  float v = (l < L_) ? x[((size_t)b * L_ + l) * D_ + d] : 0.0f;
  xp[i] = (__bf16)v;
}

// ---------------------------------------------------------------------------
// Gather window centers (pos = 5n+5) into bf16 [B, NWP_, D]; pad rows zero.
// ---------------------------------------------------------------------------
__global__ void build_xcenters(const float* __restrict__ x, __bf16* __restrict__ xc) {
  size_t i = (size_t)blockIdx.x * blockDim.x + threadIdx.x;
  const size_t total = (size_t)B_ * NWP_ * D_;
  if (i >= total) return;
  int d = (int)(i & (D_ - 1));
  size_t row = i >> 10;
  int n = (int)(row % NWP_);
  int b = (int)(row / NWP_);
  float v = (n < NW_) ? x[((size_t)b * L_ + (W_ * n + W_)) * D_ + d] : 0.0f;
  xc[i] = (__bf16)v;
}

// ---------------------------------------------------------------------------
// GEMM: C[M,N] = A[M,K](bf16,row-major) * Wt[N,K](bf16,row-major)^T + bias.
// 256 threads = 8 waves; wave tile 16(M) x 64(N); block tile 128 x 64.
// A/B fragments use contiguous K order (lane half h holds K 16h..16h+15).  K in steps of 32,
// unrolled x2 with ping-pong buffers (no rotation copies).
// ---------------------------------------------------------------------------
template <bool OUT_BF16>
__global__ __launch_bounds__(256)
void gemm_bf16_wmma(const __bf16* __restrict__ A,
                    const __bf16* __restrict__ Wt,
                    const float*  __restrict__ bias,
                    void* __restrict__ Cout,
                    int M, int N, int K) {
  const int tid  = threadIdx.x;
  const int wv   = tid >> 5;    // wave id 0..7 -> M sub-tile
  const int lane = tid & 31;
  const int half = lane >> 4;   // 0: lanes 0-15, 1: lanes 16-31
  const int lr   = lane & 15;

  const int mbase = blockIdx.y * 128 + wv * 16;
  const int nbase = blockIdx.x * 64;

  v8f acc[4] = {};

  // A fragment 16x32: lanes 0-15 hold K 0-15;
  // lanes 16-31 hold K 16-31 (matches B).  aptr folds in the half offset.
  const __bf16* aptr = A + (size_t)(mbase + lr) * K + half * 16;
  // B fragment 32x16: lane lr holds column (nbase+j*16+lr); halves hold
  // K 0-15 / 16-31.  Weights are N x K row-major == B^T -> 32B contiguous.
  const __bf16* wp0 = Wt + (size_t)(nbase + 0  + lr) * K + half * 16;
  const __bf16* wp1 = Wt + (size_t)(nbase + 16 + lr) * K + half * 16;
  const __bf16* wp2 = Wt + (size_t)(nbase + 32 + lr) * K + half * 16;
  const __bf16* wp3 = Wt + (size_t)(nbase + 48 + lr) * K + half * 16;

  struct Frag { v8bf a0, a1; v16bf b0, b1, b2, b3; };

  auto load_frag = [&](int k) {
    Frag f;
    f.a0 = *(const v8bf*)(aptr + k);
    f.a1 = *(const v8bf*)(aptr + k + 8);
    f.b0 = *(const v16bf*)(wp0 + k);
    f.b1 = *(const v16bf*)(wp1 + k);
    f.b2 = *(const v16bf*)(wp2 + k);
    f.b3 = *(const v16bf*)(wp3 + k);
    return f;
  };

  auto compute = [&](const Frag& f) {
    v16bf afrag;
#pragma unroll
    for (int i = 0; i < 8; ++i) { afrag[i] = f.a0[i]; afrag[i + 8] = f.a1[i]; }
    acc[0] = __builtin_amdgcn_wmma_f32_16x16x32_bf16(
        false, afrag, false, f.b0, (short)0, acc[0], false, false);
    acc[1] = __builtin_amdgcn_wmma_f32_16x16x32_bf16(
        false, afrag, false, f.b1, (short)0, acc[1], false, false);
    acc[2] = __builtin_amdgcn_wmma_f32_16x16x32_bf16(
        false, afrag, false, f.b2, (short)0, acc[2], false, false);
    acc[3] = __builtin_amdgcn_wmma_f32_16x16x32_bf16(
        false, afrag, false, f.b3, (short)0, acc[3], false, false);
  };

  // ---- ping-pong pipelined K loop (K multiple of 64) ----------------------
  Frag f0 = load_frag(0);
  int k0 = 0;
  for (; k0 < K - 64; k0 += 64) {
    Frag f1 = load_frag(k0 + 32);
    __builtin_prefetch((const void*)(aptr + k0 + 128), 0, 3);  // WGP scope
    compute(f0);
    f0 = load_frag(k0 + 64);
    compute(f1);
  }
  {
    Frag f1 = load_frag(k0 + 32);
    compute(f0);
    compute(f1);
  }

  // C/D layout (ISA 7.12.2): VGPR r, lanes 0-15 -> row mbase+r, col lr;
  // lanes 16-31 -> row mbase+8+r.
#pragma unroll
  for (int j = 0; j < 4; ++j) {
    const int col = nbase + j * 16 + lr;
    const float bv = bias ? bias[col] : 0.0f;
#pragma unroll
    for (int r = 0; r < 8; ++r) {
      const int row = mbase + r + half * 8;
      const float v = acc[j][r] + bv;
      if (OUT_BF16)
        ((__bf16*)Cout)[(size_t)row * N + col] = (__bf16)v;
      else
        ((float*)Cout)[(size_t)row * N + col] = v;
    }
  }
}

// ---------------------------------------------------------------------------
// Windowed attention: one block per (b, n), one wave per head.
// scores[t] = (q . k_t)/sqrt(hd) over T=11 taps; softmax; o = sum p_t v_t.
// Each lane owns 4 of the 128 head dims; shuffle butterfly reductions.
// ---------------------------------------------------------------------------
__global__ __launch_bounds__(256)
void win_attention(const float* __restrict__ Q,
                   const __bf16* __restrict__ Kp,
                   const __bf16* __restrict__ Vp,
                   __bf16* __restrict__ O) {
  const int bn = blockIdx.x;
  const int b = bn / NWP_;
  const int n = bn % NWP_;
  const int h = threadIdx.x >> 5;
  const int lane = threadIdx.x & 31;

  const size_t obase = (size_t)bn * D_ + h * HD_ + lane * 4;
  if (n >= NW_) {                 // pad rows: zero so out_proj GEMM is clean
    v4bf z = {};
    *(v4bf*)(O + obase) = z;
    return;
  }

  float q[4];
  const size_t qoff = (size_t)bn * D_ + h * HD_ + lane * 4;
#pragma unroll
  for (int i = 0; i < 4; ++i) q[i] = Q[qoff + i];

  const float scale = 0.08838834764831845f;  // 1/sqrt(128)
  const size_t kbase = ((size_t)b * LP_ + W_ * n) * D_ + h * HD_ + lane * 4;

  float s[T_];
#pragma unroll
  for (int t = 0; t < T_; ++t) {
    v4bf kv = *(const v4bf*)(Kp + kbase + (size_t)t * D_);
    float p = q[0] * (float)kv[0] + q[1] * (float)kv[1] +
              q[2] * (float)kv[2] + q[3] * (float)kv[3];
#pragma unroll
    for (int off = 16; off >= 1; off >>= 1) p += __shfl_xor(p, off, 32);
    s[t] = p * scale;
  }

  float m = s[0];
#pragma unroll
  for (int t = 1; t < T_; ++t) m = fmaxf(m, s[t]);
  float sum = 0.0f;
#pragma unroll
  for (int t = 0; t < T_; ++t) { s[t] = __expf(s[t] - m); sum += s[t]; }
  const float inv = 1.0f / sum;

  float o[4] = {0.f, 0.f, 0.f, 0.f};
#pragma unroll
  for (int t = 0; t < T_; ++t) {
    v4bf vv = *(const v4bf*)(Vp + kbase + (size_t)t * D_);
#pragma unroll
    for (int i = 0; i < 4; ++i) o[i] += s[t] * (float)vv[i];
  }
  v4bf ov;
#pragma unroll
  for (int i = 0; i < 4; ++i) ov[i] = (__bf16)(o[i] * inv);
  *(v4bf*)(O + obase) = ov;
}

// ---------------------------------------------------------------------------
// Baseline: log_softmax of a zero row with nC=2 is -ln(2) everywhere.
// ---------------------------------------------------------------------------
__global__ void fill_base(float* __restrict__ out, size_t n) {
  size_t i = (size_t)blockIdx.x * blockDim.x + threadIdx.x;
  if (i < n) out[i] = -0.6931471805599453f;
}

// ---------------------------------------------------------------------------
// Classifier + log_softmax scatter: one wave per (b, n) window.
// ---------------------------------------------------------------------------
__global__ __launch_bounds__(256)
void probs_scatter(const float* __restrict__ feat,
                   const float* __restrict__ ow,
                   const float* __restrict__ ob,
                   float* __restrict__ out) {
  const int wid = blockIdx.x * 8 + (threadIdx.x >> 5);
  const int lane = threadIdx.x & 31;
  if (wid >= B_ * NW_) return;
  const int b = wid / NW_;
  const int n = wid % NW_;

  const float* fr = feat + (size_t)(b * NWP_ + n) * D_;
  float s0 = 0.f, s1 = 0.f;
  for (int d = lane; d < D_; d += 32) {
    const float f = fr[d];
    s0 += f * ow[d];
    s1 += f * ow[D_ + d];
  }
#pragma unroll
  for (int off = 16; off >= 1; off >>= 1) {
    s0 += __shfl_xor(s0, off, 32);
    s1 += __shfl_xor(s1, off, 32);
  }
  if (lane == 0) {
    s0 += ob[0];
    s1 += ob[1];
    const float m = fmaxf(s0, s1);
    const float lse = m + __logf(__expf(s0 - m) + __expf(s1 - m));
    const size_t o = ((size_t)b * L_ + W_ * n) * NC_;
    out[o] = s0 - lse;
    out[o + 1] = s1 - lse;
  }
}

// ---------------------------------------------------------------------------
extern "C" void kernel_launch(void* const* d_in, const int* in_sizes, int n_in,
                              void* d_out, int out_size, void* d_ws,
                              size_t ws_size, hipStream_t stream) {
  (void)in_sizes; (void)n_in; (void)out_size; (void)ws_size;
  const float* x   = (const float*)d_in[0];
  const float* ipw = (const float*)d_in[1];  // (3D, D): wq | wk | wv
  const float* ipb = (const float*)d_in[2];  // (3D,)
  const float* opw = (const float*)d_in[3];  // (D, D)
  const float* opb = (const float*)d_in[4];  // (D,)
  const float* ow  = (const float*)d_in[5];  // (nC, D)
  const float* ob  = (const float*)d_in[6];  // (nC,)
  // d_in[7] = x_len: unused by the reference computation.
  float* out = (float*)d_out;

  // ---- workspace carve-out -------------------------------------------------
  char* ws = (char*)d_ws;
  size_t off = 0;
  auto carve = [&](size_t bytes) -> void* {
    void* p = ws + off;
    off += (bytes + 255) & ~(size_t)255;
    return p;
  };
  __bf16* wqkv = (__bf16*)carve((size_t)3 * D_ * D_ * 2);
  __bf16* opwb = (__bf16*)carve((size_t)D_ * D_ * 2);
  __bf16* xpad = (__bf16*)carve((size_t)B_ * LP_ * D_ * 2);
  __bf16* xc   = (__bf16*)carve((size_t)B_ * NWP_ * D_ * 2);
  __bf16* kp   = (__bf16*)carve((size_t)B_ * LP_ * D_ * 2);
  __bf16* vp   = (__bf16*)carve((size_t)B_ * LP_ * D_ * 2);
  float*  qf   = (float*) carve((size_t)B_ * NWP_ * D_ * 4);
  __bf16* obuf = (__bf16*)carve((size_t)B_ * NWP_ * D_ * 2);
  float*  feat = (float*) carve((size_t)B_ * NWP_ * D_ * 4);

  // ---- 1. casts / gathers --------------------------------------------------
  {
    size_t n = (size_t)3 * D_ * D_;
    cvt_f32_bf16<<<(unsigned)((n + 255) / 256), 256, 0, stream>>>(ipw, wqkv, n);
  }
  {
    size_t n = (size_t)D_ * D_;
    cvt_f32_bf16<<<(unsigned)((n + 255) / 256), 256, 0, stream>>>(opw, opwb, n);
  }
  {
    size_t n = (size_t)B_ * LP_ * D_;
    build_xpad<<<(unsigned)((n + 255) / 256), 256, 0, stream>>>(x, xpad);
  }
  {
    size_t n = (size_t)B_ * NWP_ * D_;
    build_xcenters<<<(unsigned)((n + 255) / 256), 256, 0, stream>>>(x, xc);
  }

  // ---- 2. projections (WMMA GEMMs) ----------------------------------------
  const int Mkv = B_ * LP_;    // 66048, multiple of 128
  const int Mq  = B_ * NWP_;   // 13312, multiple of 128
  dim3 blk(256);
  dim3 grid_kv(D_ / 64, Mkv / 128);
  dim3 grid_q (D_ / 64, Mq  / 128);

  gemm_bf16_wmma<true ><<<grid_kv, blk, 0, stream>>>(
      xpad, wqkv + (size_t)D_ * D_,     ipb + D_,     (void*)kp, Mkv, D_, D_);
  gemm_bf16_wmma<true ><<<grid_kv, blk, 0, stream>>>(
      xpad, wqkv + (size_t)2 * D_ * D_, ipb + 2 * D_, (void*)vp, Mkv, D_, D_);
  gemm_bf16_wmma<false><<<grid_q, blk, 0, stream>>>(
      xc,   wqkv,                        ipb,          (void*)qf, Mq, D_, D_);

  // ---- 3. attention --------------------------------------------------------
  win_attention<<<B_ * NWP_, 256, 0, stream>>>(qf, kp, vp, obuf);

  // ---- 4. output projection (WMMA GEMM) -----------------------------------
  gemm_bf16_wmma<false><<<grid_q, blk, 0, stream>>>(
      obuf, opwb, opb, (void*)feat, Mq, D_, D_);

  // ---- 5. classifier + log_softmax scatter --------------------------------
  {
    size_t n = (size_t)B_ * L_ * NC_;
    fill_base<<<(unsigned)((n + 255) / 256), 256, 0, stream>>>(out, n);
  }
  probs_scatter<<<(B_ * NW_ + 7) / 8, 256, 0, stream>>>(feat, ow, ob, out);
}
